// RatMultiHeadedAttention_31018253812379
// MI455X (gfx1250) — matrix operand order repaired and checked by reading, not compiled, not run on hardware
//
#include <hip/hip_runtime.h>
#include <hip/hip_bf16.h>

typedef __attribute__((ext_vector_type(16))) _Float16 v16h;
typedef __attribute__((ext_vector_type(8)))  float    v8f;
typedef unsigned int u32x4 __attribute__((ext_vector_type(4)));
typedef int          i32x4 __attribute__((ext_vector_type(4)));
typedef int          i32x8 __attribute__((ext_vector_type(8)));

#define B_   4
#define S_   512
#define HID_ 1024
#define H_   16
#define D_   64
#define SJ   32
#define NJB  (S_ / SJ)               // 16 j-blocks
#define RELBUF_BYTES (SJ * 16 * D_ * 4)   // 131072 B per f32 rel tile

#if __has_builtin(__builtin_amdgcn_tensor_load_to_lds)
#define HAVE_TDM 1
#else
#define HAVE_TDM 0
#endif

// A-fragment K offset within a 32-wide K chunk (16-bit A, 16x32):
// lanes 0-15 hold K 0..15, lanes 16-31 hold K 16..31
__device__ __forceinline__ int a_koff(int e, int half) {
    return e + 16 * half;
}
// B-fragment K offset (16-bit B, 32x16): lanes 0-15 K 0..15, lanes 16-31 K 16..31
__device__ __forceinline__ int b_koff(int e, int half) {
    return e + 16 * half;
}

// LDS byte offset of a generic pointer that points into LDS (addrspacecast to AS3)
__device__ __forceinline__ unsigned lds_addr_of(const void* p) {
    return (unsigned)(unsigned long long)(const __attribute__((address_space(3))) char*)p;
}

#if HAVE_TDM
// Issue one TDM load of a 3D rel tile: [j'=32][i'=16][d=64] f32 -> LDS (contiguous).
// dim0 = d (64 elems, stride 1), dim1 = i' (stride D_), dim2 = j' (stride S_*D_).
__device__ __forceinline__ void tdm_load_rel_tile(const float* gbase, unsigned lds_byte_off)
{
    unsigned long long ga = (unsigned long long)(size_t)gbase;
    u32x4 g0;
    g0[0] = 1u;                                               // count=1 (valid user D#)
    g0[1] = lds_byte_off;                                     // lds_addr (bytes)
    g0[2] = (unsigned)(ga & 0xffffffffu);                     // global_addr[31:0]
    g0[3] = (unsigned)((ga >> 32) & 0x1ffffffu) | (2u << 30); // global_addr[56:32] | type=2
    i32x8 g1;
    g1[0] = (int)(2u << 16);                                  // data_size=2 (4 bytes), mask=0
    g1[1] = (int)(64u << 16);                                 // tensor_dim0 = 64  (bits 79:48)
    g1[2] = (int)(512u << 16);                                // tensor_dim1 = 512 (bits 111:80)
    g1[3] = (int)(64u << 16);                                 // tile_dim0 = 64    (bits 127:112)
    g1[4] = (int)(16u | (32u << 16));                         // tile_dim1=16, tile_dim2=32
    g1[5] = 64;                                               // tensor_dim0_stride = 64
    g1[6] = (int)(0x8000u << 16);                             // tensor_dim1_stride = 32768 (lo16)
    g1[7] = 0;                                                // tensor_dim1_stride (hi32) = 0
    i32x4 g2;
    g2[0] = 512;                                              // tensor_dim2
    g2[1] = 0;                                                // tensor_dim3
    g2[2] = 0;                                                // tensor_dim2_stride lo
    g2[3] = 0;                                                // stride hi | tile_dim3 = 0
    i32x4 g3 = {0, 0, 0, 0};
    i32x8 g4 = {0, 0, 0, 0, 0, 0, 0, 0};                      // extra group (zero-filled)
    __builtin_amdgcn_tensor_load_to_lds(g0, g1, g2, g3, g4, 0);
}
#endif

// ---------------------------------------------------------------------------
// Kernel 1: Y(b,h,s,d) f16 = (X(2048x1024) @ W^T + bias) * scale, head-split.
// ---------------------------------------------------------------------------
__global__ __launch_bounds__(256) void proj_qkv_kernel(
    const float* __restrict__ X, const float* __restrict__ W,
    const float* __restrict__ bias, _Float16* __restrict__ Y, float scale)
{
    const int lane = threadIdx.x & 31;
    const int wv   = threadIdx.x >> 5;
    const int tile = blockIdx.x * 8 + wv;       // 8192 tiles total
    const int tm   = tile >> 6;
    const int tn   = tile & 63;
    const int half = lane >> 4, m = lane & 15;
    const int rowA = tm * 16 + m;
    const int colB = tn * 16 + m;

    v8f c = {};
    for (int k0 = 0; k0 < HID_; k0 += 32) {
        v16h a, b;
        #pragma unroll
        for (int e = 0; e < 16; ++e) {
            a[e] = (_Float16)X[(size_t)rowA * HID_ + k0 + a_koff(e, half)];
            b[e] = (_Float16)W[(size_t)colB * HID_ + k0 + b_koff(e, half)];
        }
        c = __builtin_amdgcn_wmma_f32_16x16x32_f16(false, a, false, b,
                                                   (short)0, c, false, false);
    }
    const int col = tn * 16 + m;
    const float bv = bias[col];
    #pragma unroll
    for (int r = 0; r < 8; ++r) {
        int row = tm * 16 + r + 8 * half;
        float v = (c[r] + bv) * scale;
        int bb = row >> 9, s = row & 511, h = col >> 6, d = col & 63;
        Y[((((size_t)bb * H_ + h) * S_) + s) * D_ + d] = (_Float16)v;
    }
}

// ---------------------------------------------------------------------------
// Kernel 2: fused RAT flash-attention. One WG (16 waves) per (b, 16-row tile).
// rel streamed via TDM (double-buffered f32 in LDS); all matmuls WMMA f16.
//   head role (wave=h): S_qk, online softmax, O_v += P*V
//   row  role (wave=i'): S_rel = (q*Kp)(16h x 64d) x rel^T, O_rel += P x rel
// ---------------------------------------------------------------------------
__global__ __launch_bounds__(512) void rat_attention_kernel(
    const _Float16* __restrict__ qh, const _Float16* __restrict__ kh,
    const _Float16* __restrict__ vh, const float* __restrict__ rel,
    const float* __restrict__ Kp,   const float* __restrict__ Vp,
    _Float16* __restrict__ ctx)
{
    extern __shared__ char smem[];
    float* s_relbuf[2];
    s_relbuf[0] = (float*)smem;                                  // [32j][16i][64d] f32 128KB
    s_relbuf[1] = (float*)(smem + RELBUF_BYTES);                 // double buffer    128KB
    float*    s_ctx   = (float*)smem;                            // epilogue alias [16i][16h][64d]
    float*    s_srel  = (float*)(smem + 2 * RELBUF_BYTES);       // [16i][16h][32j] 32KB
    _Float16* s_p     = (_Float16*)(smem + 2 * RELBUF_BYTES + 32768);   // [16h][16i][32j] 16KB
    float*    s_scale = (float*)(smem + 2 * RELBUF_BYTES + 32768 + 16384);
    float*    s_l     = (float*)(smem + 2 * RELBUF_BYTES + 32768 + 16384 + 1024);

    const int lane  = threadIdx.x & 31;
    const int w     = threadIdx.x >> 5;   // head id (head role) / row id (row role)
    const int half  = lane >> 4;
    const int mlane = lane & 15;
    const int bb    = blockIdx.x >> 5;    // 32 i-tiles per batch
    const int i0    = (blockIdx.x & 31) * 16;

    // Preload A fragments: q (head role) and q*Kp (row role), K = 64 (2 chunks)
    v16h aq[2], aqk[2];
    #pragma unroll
    for (int kc = 0; kc < 2; ++kc) {
        #pragma unroll
        for (int e = 0; e < 16; ++e) {
            int kk = kc * 32 + a_koff(e, half);
            aq[kc][e] = qh[((((size_t)bb * H_ + w) * S_) + (i0 + mlane)) * D_ + kk];
            float qv = (float)qh[((((size_t)bb * H_ + mlane) * S_) + (i0 + w)) * D_ + kk];
            aqk[kc][e] = (_Float16)(qv * Kp[mlane * D_ + kk]);
        }
    }

    v8f ov[4] = {{0}}, orel[4] = {{0}};
    float mrow[8], lrow[8];
    #pragma unroll
    for (int r = 0; r < 8; ++r) { mrow[r] = -1e30f; lrow[r] = 0.f; }

#if HAVE_TDM
    // kick the DMA for the first rel tile before entering the loop
    if (w == 0) {
        tdm_load_rel_tile(&rel[(((size_t)bb * S_ + 0) * S_ + i0) * D_], lds_addr_of(s_relbuf[0]));
    }
#endif

    for (int jb = 0; jb < NJB; ++jb) {
        const int j0 = jb * SJ;
        float* s_cur = s_relbuf[jb & 1];

#if HAVE_TDM
        if (w == 0) __builtin_amdgcn_s_wait_tensorcnt(0);   // tile jb landed (wave 0 tracks TDM)
        __syncthreads();   // release tile jb to all waves; all done reading the other buffer
        if (w == 0 && jb + 1 < NJB) {
            tdm_load_rel_tile(&rel[(((size_t)bb * S_ + (j0 + SJ)) * S_ + i0) * D_],
                              lds_addr_of(s_relbuf[(jb + 1) & 1]));
        }
#else
        __syncthreads();
        {   // cooperative fallback: thread -> (j', i'), 64 contiguous d's
            int jj = threadIdx.x >> 4, ii = threadIdx.x & 15;
            const float4* src = (const float4*)&rel[((((size_t)bb * S_) + (j0 + jj)) * S_ + (i0 + ii)) * D_];
            float4* dst = (float4*)&s_cur[(jj * 16 + ii) * D_];
            #pragma unroll
            for (int d4 = 0; d4 < 16; ++d4) dst[d4] = src[d4];
        }
        __syncthreads();
#endif

        // row role: S_rel[i'=w] (16h x 32j) = aqk (16h x 64d) x rel^T (64d x 32j)
        #pragma unroll
        for (int jt = 0; jt < 2; ++jt) {
            v8f c = {};
            #pragma unroll
            for (int kc = 0; kc < 2; ++kc) {
                v16h bf;
                #pragma unroll
                for (int e = 0; e < 16; ++e) {
                    int d  = kc * 32 + b_koff(e, half);
                    int jp = jt * 16 + mlane;
                    bf[e] = (_Float16)s_cur[(jp * 16 + w) * D_ + d];
                }
                c = __builtin_amdgcn_wmma_f32_16x16x32_f16(false, aqk[kc], false, bf,
                                                           (short)0, c, false, false);
            }
            #pragma unroll
            for (int r = 0; r < 8; ++r)
                s_srel[(w * 16 + (r + 8 * half)) * 32 + jt * 16 + mlane] = c[r];
        }

        // head role: S_qk (16i x 32j) = q (16i x 64d) x k^T (64d x 32j)
        v8f sA = {}, sB = {};
        #pragma unroll
        for (int jt = 0; jt < 2; ++jt) {
            v8f c = {};
            #pragma unroll
            for (int kc = 0; kc < 2; ++kc) {
                v16h bf;
                #pragma unroll
                for (int e = 0; e < 16; ++e) {
                    int d = kc * 32 + b_koff(e, half);
                    bf[e] = kh[((((size_t)bb * H_ + w) * S_) + (j0 + jt * 16 + mlane)) * D_ + d];
                }
                c = __builtin_amdgcn_wmma_f32_16x16x32_f16(false, aq[kc], false, bf,
                                                           (short)0, c, false, false);
            }
            if (jt == 0) sA = c; else sB = c;
        }
        __syncthreads();

        // combine (transposed exchange) + online softmax
        float scl[8];
        #pragma unroll
        for (int r = 0; r < 8; ++r) {
            int row = r + 8 * half;
            sA[r] += s_srel[(row * 16 + w) * 32 + mlane];
            sB[r] += s_srel[(row * 16 + w) * 32 + 16 + mlane];
            float t = fmaxf(sA[r], sB[r]);
            #pragma unroll
            for (int o = 1; o < 16; o <<= 1) t = fmaxf(t, __shfl_xor(t, o, 32));
            float nm = fmaxf(mrow[r], t);
            scl[r] = __expf(mrow[r] - nm);
            sA[r] = __expf(sA[r] - nm);
            sB[r] = __expf(sB[r] - nm);
            float rs = sA[r] + sB[r];
            #pragma unroll
            for (int o = 1; o < 16; o <<= 1) rs += __shfl_xor(rs, o, 32);
            lrow[r] = lrow[r] * scl[r] + rs;
            mrow[r] = nm;
        }
        // publish P (f16) and row rescale factors; rescale O_v
        #pragma unroll
        for (int r = 0; r < 8; ++r) {
            int row = r + 8 * half;
            s_p[(w * 16 + row) * 32 + mlane]      = (_Float16)sA[r];
            s_p[(w * 16 + row) * 32 + 16 + mlane] = (_Float16)sB[r];
        }
        if (mlane == 0) {
            #pragma unroll
            for (int r = 0; r < 8; ++r) s_scale[w * 16 + 8 * half + r] = scl[r];
        }
        #pragma unroll
        for (int dt = 0; dt < 4; ++dt)
            #pragma unroll
            for (int r = 0; r < 8; ++r) ov[dt][r] *= scl[r];
        __syncthreads();

        // head role: O_v += P (16i x 32j) x V (32j x 64d)
        {
            v16h ap;
            #pragma unroll
            for (int e = 0; e < 16; ++e)
                ap[e] = s_p[(w * 16 + mlane) * 32 + a_koff(e, half)];
            #pragma unroll
            for (int dt = 0; dt < 4; ++dt) {
                v16h bf;
                #pragma unroll
                for (int e = 0; e < 16; ++e) {
                    int k = b_koff(e, half);
                    bf[e] = vh[((((size_t)bb * H_ + w) * S_) + (j0 + k)) * D_ + dt * 16 + mlane];
                }
                ov[dt] = __builtin_amdgcn_wmma_f32_16x16x32_f16(false, ap, false, bf,
                                                                (short)0, ov[dt], false, false);
            }
        }
        // row role: rescale O_rel rows, then O_rel += P_slice x rel (32j x 64d at i'=w)
        {
            #pragma unroll
            for (int r = 0; r < 8; ++r) {
                float f = s_scale[(r + 8 * half) * 16 + w];
                #pragma unroll
                for (int dt = 0; dt < 4; ++dt) orel[dt][r] *= f;
            }
            v16h ap;
            #pragma unroll
            for (int e = 0; e < 16; ++e)
                ap[e] = s_p[(mlane * 16 + w) * 32 + a_koff(e, half)];
            #pragma unroll
            for (int dt = 0; dt < 4; ++dt) {
                v16h bf;
                #pragma unroll
                for (int e = 0; e < 16; ++e) {
                    int k = b_koff(e, half);
                    bf[e] = (_Float16)s_cur[(k * 16 + w) * D_ + dt * 16 + mlane];
                }
                orel[dt] = __builtin_amdgcn_wmma_f32_16x16x32_f16(false, ap, false, bf,
                                                                  (short)0, orel[dt], false, false);
            }
        }
    }

    // ---- epilogue: publish l, finalize rel part, combine, store ctx (f16)
    __syncthreads();
    if (mlane == 0) {
        #pragma unroll
        for (int r = 0; r < 8; ++r) s_l[w * 16 + 8 * half + r] = lrow[r];
    }
    __syncthreads();
    #pragma unroll
    for (int dt = 0; dt < 4; ++dt) {
        #pragma unroll
        for (int r = 0; r < 8; ++r) {
            int h = r + 8 * half;
            int d = dt * 16 + mlane;
            s_ctx[(w * 16 + h) * D_ + d] = orel[dt][r] * Vp[h * D_ + d] / s_l[h * 16 + w];
        }
    }
    __syncthreads();
    #pragma unroll
    for (int dt = 0; dt < 4; ++dt) {
        #pragma unroll
        for (int r = 0; r < 8; ++r) {
            int row = r + 8 * half;
            int d = dt * 16 + mlane;
            float v = ov[dt][r] / lrow[r] + s_ctx[(row * 16 + w) * D_ + d];
            ctx[((((size_t)bb * S_) + (i0 + row)) * H_ + w) * D_ + d] = (_Float16)v;
        }
    }
}

// ---------------------------------------------------------------------------
// Kernel 3: out(2048x1024) f32 = ctx(f16) @ Wo^T + bo
// ---------------------------------------------------------------------------
__global__ __launch_bounds__(256) void out_proj_kernel(
    const _Float16* __restrict__ Xh, const float* __restrict__ W,
    const float* __restrict__ bias, float* __restrict__ Y)
{
    const int lane = threadIdx.x & 31;
    const int wv   = threadIdx.x >> 5;
    const int tile = blockIdx.x * 8 + wv;
    const int tm   = tile >> 6;
    const int tn   = tile & 63;
    const int half = lane >> 4, m = lane & 15;
    const int rowA = tm * 16 + m;
    const int colB = tn * 16 + m;

    v8f c = {};
    for (int k0 = 0; k0 < HID_; k0 += 32) {
        v16h a, b;
        #pragma unroll
        for (int e = 0; e < 16; ++e) {
            a[e] = Xh[(size_t)rowA * HID_ + k0 + a_koff(e, half)];
            b[e] = (_Float16)W[(size_t)colB * HID_ + k0 + b_koff(e, half)];
        }
        c = __builtin_amdgcn_wmma_f32_16x16x32_f16(false, a, false, b,
                                                   (short)0, c, false, false);
    }
    const int col = tn * 16 + m;
    const float bv = bias[col];
    #pragma unroll
    for (int r = 0; r < 8; ++r) {
        int row = tm * 16 + r + 8 * half;
        Y[(size_t)row * HID_ + col] = c[r] + bv;
    }
}

// ---------------------------------------------------------------------------
extern "C" void kernel_launch(void* const* d_in, const int* in_sizes, int n_in,
                              void* d_out, int out_size, void* d_ws, size_t ws_size,
                              hipStream_t stream) {
    const float* key_  = (const float*)d_in[0];
    const float* value = (const float*)d_in[1];
    const float* query = (const float*)d_in[2];
    const float* rel   = (const float*)d_in[3];
    // d_in[4] = dropout (unused)
    const float* Wq = (const float*)d_in[5];
    const float* bq = (const float*)d_in[6];
    const float* Wk = (const float*)d_in[7];
    const float* bk = (const float*)d_in[8];
    const float* Wv = (const float*)d_in[9];
    const float* bv = (const float*)d_in[10];
    const float* Wo = (const float*)d_in[11];
    const float* bo = (const float*)d_in[12];
    const float* Kp = (const float*)d_in[13];
    const float* Vp = (const float*)d_in[14];

    const size_t NTOK = (size_t)B_ * S_ * HID_;  // 2,097,152
    _Float16* qh  = (_Float16*)d_ws;
    _Float16* kh  = qh + NTOK;
    _Float16* vh  = kh + NTOK;
    _Float16* ctx = vh + NTOK;                   // 16 MB total

    proj_qkv_kernel<<<1024, 256, 0, stream>>>(query, Wq, bq, qh, 0.125f);
    proj_qkv_kernel<<<1024, 256, 0, stream>>>(key_,  Wk, bk, kh, 1.0f);
    proj_qkv_kernel<<<1024, 256, 0, stream>>>(value, Wv, bv, vh, 1.0f);

    // fused attention: 128 WGs x 512 threads; 306 KB dynamic LDS
    size_t shmem = 2 * RELBUF_BYTES + 32768 + 16384 + 1024 + 1024;
    rat_attention_kernel<<<B_ * (S_ / 16), 512, shmem, stream>>>(
        qh, kh, vh, rel, Kp, Vp, ctx);

    out_proj_kernel<<<1024, 256, 0, stream>>>(ctx, Wo, bo, (float*)d_out);
}
